// TimeSeriesRNN_57131654972194
// MI455X (gfx1250) — matrix operand order repaired and checked by reading, not compiled, not run on hardware
//
#include <hip/hip_runtime.h>

// ---------------------------------------------------------------------------
// Types for CDNA5 WMMA (wave32, 16x16x32 bf16 -> f32)
// ---------------------------------------------------------------------------
typedef __bf16 bf16;
typedef __bf16 v16bf __attribute__((ext_vector_type(16)));
typedef float  v8f   __attribute__((ext_vector_type(8)));

struct Frag32 { uint4 lo; uint4 hi; };

__device__ __forceinline__ v16bf load_frag(const bf16* p0, const bf16* p1) {
  Frag32 f;
  f.lo = *reinterpret_cast<const uint4*>(p0);
  f.hi = *reinterpret_cast<const uint4*>(p1);
  return __builtin_bit_cast(v16bf, f);
}

#define NWG 256u

// ---------------------------------------------------------------------------
// Grid-wide barrier: monotonic counter, re-zeroed by init kernel each launch.
// ---------------------------------------------------------------------------
__device__ __forceinline__ void gsync(unsigned* ctr, unsigned& gen) {
  __threadfence();
  __syncthreads();
  gen += 1;                        // uniform across all threads
  if (threadIdx.x == 0) {
    unsigned target = gen * NWG;
    __hip_atomic_fetch_add(ctr, 1u, __ATOMIC_ACQ_REL, __HIP_MEMORY_SCOPE_AGENT);
    while (__hip_atomic_load(ctr, __ATOMIC_ACQUIRE, __HIP_MEMORY_SCOPE_AGENT) < target) {
      __builtin_amdgcn_s_sleep(2);
    }
  }
  __syncthreads();
  __threadfence();
}

// ---------------------------------------------------------------------------
// One LSTM layer step for this WG's (m, j) tile.
//   A (64 x K) = [ x-part (nxkt k-tiles, row stride 64) |
//                  srcA (32 k-tiles, row stride 1024)   |
//                  srcB (rest, row stride 1024) ]
//   Wave w computes gate tile w (N-tile = w*64 + j) via WMMA over K.
//   Gates staged in LDS; fused cell update writes c (f32) and h (bf16).
// ---------------------------------------------------------------------------
__device__ __forceinline__ void layer_step(
    int wave, int lane, int m, int j,
    const bf16* __restrict__ xpart, int nxkt,
    const bf16* __restrict__ srcA, const bf16* __restrict__ srcB,
    int nkt, const bf16* __restrict__ swzW,
    const float* __restrict__ bias, float* __restrict__ C,
    bf16* __restrict__ Hout, float (*g_lds)[16][16])
{
  const int r15 = lane & 15;
  const int hi8 = (lane >> 4) * 8;
  const int row = m * 16 + r15;
  const int jt  = wave * 64 + j;                     // gate N-tile index (0..255)
  const bf16* __restrict__ bbase = swzW + (((size_t)jt * nkt) << 9) + (lane << 4);

  v8f acc = {};
  for (int kt = 0; kt < nkt; ++kt) {
    const bf16* ab;
    if (kt < nxkt) {
      ab = xpart + (size_t)row * 64 + kt * 32;
    } else {
      int kh = kt - nxkt;
      ab = (kh < 32) ? srcA + (size_t)row * 1024 + kh * 32
                     : srcB + (size_t)row * 1024 + (kh - 32) * 32;
    }
    v16bf a = load_frag(ab + hi8, ab + 16 + hi8);
    const bf16* bb = bbase + ((size_t)kt << 9);
    __builtin_prefetch(bb + 512, 0, 0);              // next 1KB weight block
    v16bf b = load_frag(bb, bb + 8);
    acc = __builtin_amdgcn_wmma_f32_16x16x32_bf16(false, a, false, b,
                                                  (short)0, acc, false, false);
  }

#pragma unroll
  for (int r = 0; r < 8; ++r) g_lds[wave][r + hi8][r15] = acc[r];
  __syncthreads();

  for (int e = threadIdx.x; e < 256; e += 128) {
    int bl = e >> 4, nc = e & 15;
    int n = j * 16 + nc;
    int b = m * 16 + bl;
    float gi = g_lds[0][bl][nc] + bias[n];
    float gf = g_lds[1][bl][nc] + bias[1024 + n];
    float gg = g_lds[2][bl][nc] + bias[2048 + n];
    float go = g_lds[3][bl][nc] + bias[3072 + n];
    float co = C[(size_t)b * 1024 + n];
    float si = 1.0f / (1.0f + __expf(-gi));
    float sf = 1.0f / (1.0f + __expf(-gf));
    float so = 1.0f / (1.0f + __expf(-go));
    float cn = sf * co + si * tanhf(gg);
    float hn = so * tanhf(cn);
    C[(size_t)b * 1024 + n] = cn;
    Hout[(size_t)b * 1024 + n] = (bf16)hn;
  }
  __syncthreads();
}

// ---------------------------------------------------------------------------
// Decoder output projection: out = h1 @ Wf^T + bf  (64x1024 @ 1024x64).
// Done by WGs 0..15 (mp = wg&3, jp = wg>>2); 4 waves split K, LDS-reduce.
// ---------------------------------------------------------------------------
__device__ __forceinline__ void proj_step(
    int wave, int lane, int wg, int t,
    const bf16* __restrict__ h1, const bf16* __restrict__ swzf,
    const float* __restrict__ bfv, float* __restrict__ out,
    bf16* __restrict__ Din, float (*g_lds)[16][16])
{
  const int mp = wg & 3, jp = wg >> 2;
  const int r15 = lane & 15, hi8 = (lane >> 4) * 8;
  const int row = mp * 16 + r15;

  v8f acc = {};
#pragma unroll
  for (int kk = 0; kk < 8; ++kk) {
    int kt = wave * 8 + kk;
    const bf16* ab = h1 + (size_t)row * 1024 + kt * 32;
    v16bf a = load_frag(ab + hi8, ab + 16 + hi8);
    const bf16* bb = swzf + (((size_t)jp * 32 + kt) << 9) + (lane << 4);
    v16bf b = load_frag(bb, bb + 8);
    acc = __builtin_amdgcn_wmma_f32_16x16x32_bf16(false, a, false, b,
                                                  (short)0, acc, false, false);
  }
#pragma unroll
  for (int r = 0; r < 8; ++r) g_lds[wave][r + hi8][r15] = acc[r];
  __syncthreads();

  for (int e = threadIdx.x; e < 256; e += 128) {
    int bl = e >> 4, vc = e & 15;
    int v = jp * 16 + vc;
    int b = mp * 16 + bl;
    float s = g_lds[0][bl][vc] + g_lds[1][bl][vc] + g_lds[2][bl][vc] +
              g_lds[3][bl][vc] + bfv[v];
    out[(size_t)b * (32 * 64) + t * 64 + v] = s;   // (B, T_OUT, V)
    Din[(size_t)b * 64 + v] = (bf16)s;             // autoregressive feedback
  }
  __syncthreads();
}

// ---------------------------------------------------------------------------
// Persistent seq2seq kernel: 256 WGs x 128 threads.
// ---------------------------------------------------------------------------
__global__ void __launch_bounds__(128, 1) lstm_seq2seq_persistent(
    const bf16* __restrict__ xbf,   // [256][64][64] bf16
    const bf16* __restrict__ swz0,  // W0 swizzled, 256x34 1KB blocks
    const bf16* __restrict__ swz1,  // W1 swizzled, 256x64 1KB blocks
    const bf16* __restrict__ swzf,  // Wf swizzled, 4x32 1KB blocks
    const float* __restrict__ b0, const float* __restrict__ b1,
    const float* __restrict__ bfv,
    bf16* H0, bf16* H1,             // double-buffered [2][64][1024]
    float* C0, float* C1,           // [64][1024]
    bf16* Din,                      // [64][64] decoder feedback
    float* out, unsigned* ctr)
{
  __shared__ float g_lds[4][16][16];
  const int wg = blockIdx.x;
  const int wave = threadIdx.x >> 5;
  const int lane = threadIdx.x & 31;
  const int m = wg & 3;
  const int j = wg >> 2;
  const size_t HS = 64 * 1024;
  unsigned gen = 0;

  // ---- Encoder: 256 steps ----
  for (int s = 0; s < 256; ++s) {
    int p = s & 1;
    layer_step(wave, lane, m, j, xbf + (size_t)s * 4096, 2,
               H0 + p * HS, H0 + p * HS, 34, swz0, b0, C0,
               H0 + (1 - p) * HS, g_lds);
    gsync(ctr, gen);
    layer_step(wave, lane, m, j, xbf, 0,
               H0 + (1 - p) * HS, H1 + p * HS, 64, swz1, b1, C1,
               H1 + (1 - p) * HS, g_lds);
    gsync(ctr, gen);
  }

  // ---- Decoder: 32 autoregressive steps ----
  for (int t = 0; t < 32; ++t) {
    int s = 256 + t;
    int p = s & 1;
    const bf16* xp = (t == 0) ? (xbf + (size_t)255 * 4096) : Din;
    layer_step(wave, lane, m, j, xp, 2,
               H0 + p * HS, H0 + p * HS, 34, swz0, b0, C0,
               H0 + (1 - p) * HS, g_lds);
    gsync(ctr, gen);
    layer_step(wave, lane, m, j, xbf, 0,
               H0 + (1 - p) * HS, H1 + p * HS, 64, swz1, b1, C1,
               H1 + (1 - p) * HS, g_lds);
    gsync(ctr, gen);
    if (wg < 16)
      proj_step(wave, lane, wg, t, H1 + (1 - p) * HS, swzf, bfv, out, Din, g_lds);
    gsync(ctr, gen);
  }
}

// ---------------------------------------------------------------------------
// Prep kernels
// ---------------------------------------------------------------------------
__global__ void zero_ws_kernel(unsigned* p, int nwords) {
  int i = blockIdx.x * blockDim.x + threadIdx.x;
  if (i < nwords) p[i] = 0u;
}

// x: (B=64, T=256, V=64) f32  ->  xbf: [T][B][V] bf16
__global__ void convert_x_kernel(const float* __restrict__ x, bf16* __restrict__ xbf) {
  int idx = blockIdx.x * blockDim.x + threadIdx.x;
  if (idx >= 256 * 64 * 64) return;
  int v = idx & 63;
  int b = (idx >> 6) & 63;
  int t = idx >> 12;
  xbf[idx] = (bf16)x[((size_t)b << 14) + (t << 6) + v];
}

// W (N x K, row-major f32) -> bf16 B-fragment blocks [jt][kt][lane][p]
// Block (jt,kt) = 1KB; lane l holds column N=jt*16+(l&15),
// K = kt*32 + (p>>3)*16 + (l>>4)*8 + (p&7)  (p = 0..15, contiguous 32B per lane).
__global__ void swizzle_w_kernel(const float* __restrict__ W, bf16* __restrict__ out,
                                 int NJT, int NKT, int K) {
  int idx = blockIdx.x * blockDim.x + threadIdx.x;
  int total = NJT * NKT * 512;
  if (idx >= total) return;
  int p   = idx & 15;
  int ln  = (idx >> 4) & 31;
  int blk = idx >> 9;
  int kt  = blk % NKT;
  int jt  = blk / NKT;
  int n = jt * 16 + (ln & 15);
  int k = kt * 32 + (p >> 3) * 16 + (ln >> 4) * 8 + (p & 7);
  out[idx] = (bf16)W[(size_t)n * K + k];
}

// ---------------------------------------------------------------------------
// Host entry
// ---------------------------------------------------------------------------
extern "C" void kernel_launch(void* const* d_in, const int* in_sizes, int n_in,
                              void* d_out, int out_size, void* d_ws, size_t ws_size,
                              hipStream_t stream) {
  (void)in_sizes; (void)n_in; (void)out_size; (void)ws_size;
  const float* x   = (const float*)d_in[0];  // (64, 256, 64)
  const float* W0  = (const float*)d_in[1];  // (4096, 1088)
  const float* b0  = (const float*)d_in[2];  // (4096,)
  const float* W1  = (const float*)d_in[3];  // (4096, 2048)
  const float* b1  = (const float*)d_in[4];  // (4096,)
  const float* Wf  = (const float*)d_in[5];  // (64, 1024)
  const float* bfv = (const float*)d_in[6];  // (64,)
  float* out = (float*)d_out;                // (64, 32, 64)

  char* w = (char*)d_ws;
  size_t o = 0;
  unsigned* ctr = (unsigned*)(w + o); o += 256;
  bf16*  H0  = (bf16*)(w + o);  o += 262144;   // [2][64][1024] bf16
  bf16*  H1  = (bf16*)(w + o);  o += 262144;
  float* C0  = (float*)(w + o); o += 262144;   // [64][1024] f32
  float* C1  = (float*)(w + o); o += 262144;
  bf16*  Din = (bf16*)(w + o);  o += 8192;     // [64][64] bf16
  bf16*  xbf = (bf16*)(w + o);  o += 2097152;  // [256][64][64] bf16
  bf16*  s0  = (bf16*)(w + o);  o += 8912896;  // W0 swizzled (256*34 KB)
  bf16*  s1  = (bf16*)(w + o);  o += 16777216; // W1 swizzled (256*64 KB)
  bf16*  sf  = (bf16*)(w + o);  o += 131072;   // Wf swizzled (4*32 KB)

  // Zero: ctr + H0 + H1 + C0 + C1 (contiguous prefix).
  const int zwords = (256 + 4 * 262144) / 4;
  zero_ws_kernel<<<(zwords + 255) / 256, 256, 0, stream>>>((unsigned*)w, zwords);

  convert_x_kernel<<<(256 * 64 * 64 + 255) / 256, 256, 0, stream>>>(x, xbf);

  swizzle_w_kernel<<<(256 * 34 * 512 + 255) / 256, 256, 0, stream>>>(W0, s0, 256, 34, 1088);
  swizzle_w_kernel<<<(256 * 64 * 512 + 255) / 256, 256, 0, stream>>>(W1, s1, 256, 64, 2048);
  swizzle_w_kernel<<<(4 * 32 * 512 + 255) / 256, 256, 0, stream>>>(Wf, sf, 4, 32, 1024);

  lstm_seq2seq_persistent<<<NWG, 128, 0, stream>>>(
      xbf, s0, s1, sf, b0, b1, bfv, H0, H1, C0, C1, Din, out, ctr);
}
